// TransformerDecoderCrossLayer_26499948216866
// MI455X (gfx1250) — matrix operand order repaired and checked by reading, not compiled, not run on hardware
//
#include <hip/hip_runtime.h>

#define DM    256
#define DM2   512
#define NHEAD 8
#define HDV   32
#define HDQK  64
#define DFFN  2048
#define NQ    512
#define HWK   4096
#define BSZ   4
#define MQ    (NQ * BSZ)   /* 2048  */
#define MK    (HWK * BSZ)  /* 16384 */
#define TBL   144          /* 3 * 2L */
#define NBIN  48           /* 2L    */

typedef __attribute__((ext_vector_type(16))) __bf16 v16bf;
typedef __attribute__((ext_vector_type(8)))  float  v8f;

union FragU { unsigned u[8]; v16bf v; };

__device__ __forceinline__ unsigned short f2bf(float x) {
  unsigned u = __float_as_uint(x);
  u += 0x7FFFu + ((u >> 16) & 1u);   // round-to-nearest-even
  return (unsigned short)(u >> 16);
}

__device__ __forceinline__ v8f wmma_bf16(v16bf a, v16bf b, v8f c) {
  return __builtin_amdgcn_wmma_f32_16x16x32_bf16(false, a, false, b, (short)0, c, false, false);
}

// ---------------------------------------------------------------------------
// f32 -> bf16 conversion
// ---------------------------------------------------------------------------
__global__ __launch_bounds__(256) void pack_bf16_kernel(const float* __restrict__ s,
                                                        unsigned short* __restrict__ d, int n) {
  int i = blockIdx.x * 256 + threadIdx.x;
  if (i < n) d[i] = f2bf(s[i]);
}

// rel tables: [NH][32][144] f32 -> [NH][144][32] bf16 (so table GEMM B is [N,K])
__global__ __launch_bounds__(256) void rel_transpose_kernel(const float* __restrict__ s,
                                                            unsigned short* __restrict__ d) {
  int i = blockIdx.x * 256 + threadIdx.x;
  if (i >= NHEAD * TBL * HDV) return;
  int c = i & 31;
  int r = i >> 5;
  int l = r % TBL;
  int h = r / TBL;
  d[(h * TBL + l) * HDV + c] = f2bf(s[(h * HDV + c) * TBL + l]);
}

// ---------------------------------------------------------------------------
// Generic batched WMMA GEMM: Y[m,n] = (sum_k A[m,k]*B[n,k] + bias[n]) * scale
// A: bf16 [M,K] lda, B: bf16 [N,K] ldb (i.e. W row-major = B^T), C: f32/bf16.
// flags: bit0 = ReLU; bits[3:2] = column remap (0 none, 1 concat-lo, 2 concat-hi)
// 8 waves/WG; wave = 16(M) x 32(N) tile; WG = 64 x 64.
// ---------------------------------------------------------------------------
__global__ __launch_bounds__(256) void gemm_kernel(
    const unsigned short* __restrict__ A, int lda, long long sA,
    const unsigned short* __restrict__ B, int ldb, long long sB,
    const float* __restrict__ bias, long long sBias,
    float* __restrict__ Cf, unsigned short* __restrict__ Cb, int ldc, long long sC,
    int M, int N, int K, float scale, int flags)
{
  const int z    = blockIdx.z;
  A += (size_t)z * sA;
  B += (size_t)z * sB;
  if (bias) bias += (size_t)z * sBias;
  const size_t coff = (size_t)z * sC;

  const int wv   = threadIdx.x >> 5;
  const int lane = threadIdx.x & 31;
  const int hi2  = lane >> 4;
  const int nc   = lane & 15;
  const int m0   = blockIdx.y * 64 + (wv >> 1) * 16;
  const int n0   = blockIdx.x * 64 + (wv & 1) * 32;

  int rA  = m0 + nc;        if (rA  >= M) rA  = M - 1;
  int rB0 = n0 + nc;        if (rB0 >= N) rB0 = N - 1;
  int rB1 = n0 + 16 + nc;   if (rB1 >= N) rB1 = N - 1;

  const unsigned short* pa  = A + (size_t)rA  * lda + hi2 * 16;  // A layout
  const unsigned short* pb0 = B + (size_t)rB0 * ldb + hi2 * 16;  // B layout
  const unsigned short* pb1 = B + (size_t)rB1 * ldb + hi2 * 16;

  v8f acc0 = {}, acc1 = {};
  for (int kk = 0; kk < K; kk += 32) {
    __builtin_prefetch(pa + kk + 64, 0, 1);
    __builtin_prefetch(pb0 + kk + 64, 0, 1);
    FragU a, b0, b1;
#pragma unroll
    for (int t = 0; t < 4; ++t) {                 // A: K pairs {2t} and {8+2t} within 16-half
      a.u[t]     = *(const unsigned*)(pa + kk + 2 * t);
      a.u[t + 4] = *(const unsigned*)(pa + kk + 8 + 2 * t);
    }
#pragma unroll
    for (int t = 0; t < 8; ++t) {                 // B: K pairs {2t} within 16-half
      b0.u[t] = *(const unsigned*)(pb0 + kk + 2 * t);
      b1.u[t] = *(const unsigned*)(pb1 + kk + 2 * t);
    }
    acc0 = wmma_bf16(a.v, b0.v, acc0);
    acc1 = wmma_bf16(a.v, b1.v, acc1);
  }

  const int cmap = flags >> 2;
#pragma unroll
  for (int v = 0; v < 8; ++v) {
    int m = m0 + v + 8 * hi2;
#pragma unroll
    for (int t = 0; t < 2; ++t) {
      int n = n0 + t * 16 + nc;
      if (m < M && n < N) {
        float x = t ? acc1[v] : acc0[v];
        if (bias) x += bias[n];
        x *= scale;
        if (flags & 1) x = fmaxf(x, 0.f);
        int dn = n;
        if (cmap == 1)      dn = (n >> 5) * 64 + (n & 31);        // content half
        else if (cmap == 2) dn = (n >> 5) * 64 + 32 + (n & 31);   // positional half
        size_t o = coff + (size_t)m * ldc + dn;
        if (Cf) Cf[o] = x;
        if (Cb) Cb[o] = f2bf(x);
      }
    }
  }
}

// ---------------------------------------------------------------------------
// Fused flash cross-attention with 3-axis RPE bias + rel-value bin histogram.
// Grid: (NQ/64, BSZ*NHEAD). Block: 128 (4 waves); each wave owns 16 query rows.
// ---------------------------------------------------------------------------
__global__ __launch_bounds__(128) void attn_kernel(
    const unsigned short* __restrict__ qh,     // [MQ, 512] bf16 (scaled)
    const unsigned short* __restrict__ kh,     // [MK, 512] bf16
    const unsigned short* __restrict__ vh,     // [MK, 256] bf16
    const float* __restrict__ sq,              // [MQ, NH, 144] f32
    const float* __restrict__ sk,              // [MK, NH, 144] f32
    const float* __restrict__ qcoord,          // [NQ, BSZ, 3]
    const float* __restrict__ kcoord,          // [HW, BSZ, 3]
    const float* __restrict__ relv,            // [NH, 32, 144] f32
    unsigned short* __restrict__ outb)         // [MQ, 256] bf16
{
  __shared__ unsigned short s_kh[32 * 64];         // 4 KB key tile
  __shared__ unsigned short s_vh[32 * 32];         // 2 KB value tile
  __shared__ float          s_kc[32][4];           // key coords
  __shared__ float          s_qc[4][16][4];        // query coords per wave
  __shared__ unsigned short s_p[4][16 * 32];       // P re-layout scratch per wave
  __shared__ float          s_w[4][16][3 * NBIN];  // rel-value histogram per wave

  const int tid  = threadIdx.x;
  const int wv   = tid >> 5;
  const int lane = tid & 31;
  const int hi2  = lane >> 4;
  const int nc   = lane & 15;
  const int bh   = blockIdx.y;
  const int b    = bh / NHEAD, h = bh % NHEAD;
  const int i0   = blockIdx.x * 64 + wv * 16;

  // Q fragments (A layout), two K=32 chunks covering HD_QK=64
  FragU aq0, aq1;
  {
    const unsigned short* p =
        qh + ((size_t)(i0 + nc) * BSZ + b) * DM2 + h * HDQK + hi2 * 16;
#pragma unroll
    for (int t = 0; t < 4; ++t) {
      aq0.u[t]     = *(const unsigned*)(p + 2 * t);
      aq0.u[t + 4] = *(const unsigned*)(p + 8 + 2 * t);
      aq1.u[t]     = *(const unsigned*)(p + 32 + 2 * t);
      aq1.u[t + 4] = *(const unsigned*)(p + 32 + 8 + 2 * t);
    }
  }
  if (lane < 16) {
#pragma unroll
    for (int a = 0; a < 3; ++a)
      s_qc[wv][lane][a] = qcoord[((size_t)(i0 + lane) * BSZ + b) * 3 + a];
  }
  {
    float* wp = &s_w[wv][0][0];
    for (int t = lane; t < 16 * 3 * NBIN; t += 32) wp[t] = 0.f;
  }

  float mrun[8], lrun[8];
#pragma unroll
  for (int v = 0; v < 8; ++v) { mrun[v] = -1e30f; lrun[v] = 0.f; }
  v8f o0 = {}, o1 = {};

  const float* sqrow = sq;  // indexed per element below

  for (int j0 = 0; j0 < HWK; j0 += 32) {
    __syncthreads();
    {  // cooperative stage of K tile (1024 u32), V tile (512 u32), key coords
      const unsigned* src = (const unsigned*)kh;
      unsigned*       dst = (unsigned*)s_kh;
#pragma unroll
      for (int t = 0; t < 8; ++t) {
        int e = tid + t * 128;
        int r = e >> 5, c = e & 31;
        dst[e] = src[((size_t)(j0 + r) * BSZ + b) * 256 + h * 32 + c];
      }
      const unsigned* srcv = (const unsigned*)vh;
      unsigned*       dstv = (unsigned*)s_vh;
#pragma unroll
      for (int t = 0; t < 4; ++t) {
        int e = tid + t * 128;
        int r = e >> 4, c = e & 15;
        dstv[e] = srcv[((size_t)(j0 + r) * BSZ + b) * 128 + h * 16 + c];
      }
      if (tid < 96) {
        int r = tid / 3, a = tid % 3;
        s_kc[r][a] = kcoord[((size_t)(j0 + r) * BSZ + b) * 3 + a];
      }
    }
    __syncthreads();

    // S = Q K^T over 32 keys -> two 16x16 tiles
    v8f s0 = {}, s1 = {};
#pragma unroll
    for (int c = 0; c < 2; ++c) {
      FragU b0, b1;
      const unsigned short* pb0 = s_kh + (size_t)nc * 64 + c * 32 + hi2 * 16;
      const unsigned short* pb1 = s_kh + (size_t)(16 + nc) * 64 + c * 32 + hi2 * 16;
#pragma unroll
      for (int t = 0; t < 8; ++t) {
        b0.u[t] = *(const unsigned*)(pb0 + 2 * t);
        b1.u[t] = *(const unsigned*)(pb1 + 2 * t);
      }
      s0 = wmma_bf16(c ? aq1.v : aq0.v, b0.v, s0);
      s1 = wmma_bf16(c ? aq1.v : aq0.v, b1.v, s1);
    }

    // RPE bias: idx from coords on the fly; gather sq/sk rows from L2
#pragma unroll
    for (int v = 0; v < 8; ++v) {
      int m  = v + 8 * hi2;
      int ig = i0 + m;
      const float* sqp = sqrow + ((size_t)ig * BSZ + b) * (NHEAD * TBL) + h * TBL;
#pragma unroll
      for (int t = 0; t < 2; ++t) {
        int jl = t * 16 + nc;
        int jg = j0 + jl;
        const float* skp = sk + ((size_t)jg * BSZ + b) * (NHEAD * TBL) + h * TBL;
        float add = 0.f;
#pragma unroll
        for (int a = 0; a < 3; ++a) {
          float rel = s_qc[wv][m][a] - s_kc[jl][a];
          int idx = (int)floorf(rel * 10.0f);
          idx = idx < -24 ? -24 : (idx > 23 ? 23 : idx);
          idx += 24;
          add += sqp[a * NBIN + idx] + skp[a * NBIN + idx];
        }
        if (t == 0) s0[v] += add; else s1[v] += add;
      }
    }

    // online softmax (rows span the 16-lane halves)
    float sf[8];
#pragma unroll
    for (int v = 0; v < 8; ++v) {
      float mx = fmaxf(s0[v], s1[v]);
#pragma unroll
      for (int d = 1; d < 16; d <<= 1) mx = fmaxf(mx, __shfl_xor(mx, d, 32));
      float mnew = fmaxf(mrun[v], mx);
      float f    = __expf(mrun[v] - mnew);
      float p0   = __expf(s0[v] - mnew);
      float p1   = __expf(s1[v] - mnew);
      float rs   = p0 + p1;
#pragma unroll
      for (int d = 1; d < 16; d <<= 1) rs += __shfl_xor(rs, d, 32);
      lrun[v] = lrun[v] * f + rs;
      mrun[v] = mnew;
      sf[v]   = f;
      o0[v] *= f; o1[v] *= f;
      s0[v] = p0; s1[v] = p1;      // S is now P (unnormalized)
    }

    // rescale + accumulate rel-value histogram (LDS atomics, per-wave region)
#pragma unroll
    for (int v = 0; v < 8; ++v) {
      int m = v + 8 * hi2;
      float f = sf[v];
      float* wr = &s_w[wv][m][0];
#pragma unroll
      for (int e = 0; e < 9; ++e) wr[nc * 9 + e] *= f;   // 144/16 lanes
    }
#pragma unroll
    for (int v = 0; v < 8; ++v) {
      int m = v + 8 * hi2;
      float* wr = &s_w[wv][m][0];
#pragma unroll
      for (int t = 0; t < 2; ++t) {
        int jl = t * 16 + nc;
        float p = t ? s1[v] : s0[v];
#pragma unroll
        for (int a = 0; a < 3; ++a) {
          float rel = s_qc[wv][m][a] - s_kc[jl][a];
          int idx = (int)floorf(rel * 10.0f);
          idx = idx < -24 ? -24 : (idx > 23 ? 23 : idx);
          atomicAdd(&wr[a * NBIN + idx + 24], p);
        }
      }
    }

    // P -> LDS -> A-fragment re-layout (wave-private scratch)
    unsigned short* pb = &s_p[wv][0];
#pragma unroll
    for (int v = 0; v < 8; ++v) {
      int m = v + 8 * hi2;
      pb[m * 32 + nc]      = f2bf(s0[v]);
      pb[m * 32 + 16 + nc] = f2bf(s1[v]);
    }
    FragU pf;
    {
      const unsigned short* pp = pb + nc * 32 + hi2 * 16;
#pragma unroll
      for (int t = 0; t < 4; ++t) {
        pf.u[t]     = *(const unsigned*)(pp + 2 * t);
        pf.u[t + 4] = *(const unsigned*)(pp + 8 + 2 * t);
      }
    }
    // V B-fragments: B^T[c][k] = s_vh[k*32 + c]
    FragU bv0, bv1;
    {
      int kb = hi2 * 16;
#pragma unroll
      for (int t = 0; t < 8; ++t) {
        int k0 = kb + 2 * t;
        unsigned l0 = s_vh[k0 * 32 + nc];
        unsigned h0 = s_vh[(k0 + 1) * 32 + nc];
        unsigned l1 = s_vh[k0 * 32 + 16 + nc];
        unsigned h1 = s_vh[(k0 + 1) * 32 + 16 + nc];
        bv0.u[t] = l0 | (h0 << 16);
        bv1.u[t] = l1 | (h1 << 16);
      }
    }
    o0 = wmma_bf16(pf.v, bv0.v, o0);
    o1 = wmma_bf16(pf.v, bv1.v, o1);
  }

  // finalize: normalize, add rel-value contribution, store bf16
#pragma unroll
  for (int v = 0; v < 8; ++v) {
    int m = v + 8 * hi2;
    float inv = 1.0f / lrun[v];
    int c0 = nc, c1 = 16 + nc;
    const float* tv = relv + (size_t)h * HDV * TBL;
    const float* w  = &s_w[wv][m][0];
    float r0 = 0.f, r1 = 0.f;
    for (int e = 0; e < 3 * NBIN; ++e) {
      float ww = w[e];
      r0 += ww * tv[c0 * TBL + e];
      r1 += ww * tv[c1 * TBL + e];
    }
    float a0 = (o0[v] + r0) * inv;
    float a1 = (o1[v] + r1) * inv;
    size_t row = (size_t)(i0 + m) * BSZ + b;
    outb[row * DM + h * HDV + c0] = f2bf(a0);
    outb[row * DM + h * HDV + c1] = f2bf(a1);
  }
}

// ---------------------------------------------------------------------------
// y = LayerNorm(xa + xb) * g + b ; one wave per 256-wide row
// ---------------------------------------------------------------------------
__global__ __launch_bounds__(256) void add_ln_kernel(
    const float* __restrict__ xa, const float* __restrict__ xb,
    const float* __restrict__ g, const float* __restrict__ be,
    float* __restrict__ of, unsigned short* __restrict__ ob, int M)
{
  int wv = threadIdx.x >> 5, lane = threadIdx.x & 31;
  int row = blockIdx.x * 8 + wv;
  if (row >= M) return;
  const float* pa = xa + (size_t)row * DM;
  const float* pb = xb + (size_t)row * DM;
  float v[8], s = 0.f;
#pragma unroll
  for (int t = 0; t < 8; ++t) { v[t] = pa[lane + t * 32] + pb[lane + t * 32]; s += v[t]; }
#pragma unroll
  for (int d = 1; d < 32; d <<= 1) s += __shfl_xor(s, d, 32);
  float mean = s * (1.f / DM);
  float vs = 0.f;
#pragma unroll
  for (int t = 0; t < 8; ++t) { float dlt = v[t] - mean; vs += dlt * dlt; }
#pragma unroll
  for (int d = 1; d < 32; d <<= 1) vs += __shfl_xor(vs, d, 32);
  float rstd = rsqrtf(vs * (1.f / DM) + 1e-5f);
#pragma unroll
  for (int t = 0; t < 8; ++t) {
    int c = lane + t * 32;
    float y = (v[t] - mean) * rstd * g[c] + be[c];
    if (of) of[(size_t)row * DM + c] = y;
    if (ob) ob[(size_t)row * DM + c] = f2bf(y);
  }
}

// ---------------------------------------------------------------------------
static void gemm(const unsigned short* A, int lda, long long sA,
                 const unsigned short* B, int ldb, long long sB,
                 const float* bias, long long sBias,
                 float* Cf, unsigned short* Cb, int ldc, long long sC,
                 int M, int N, int K, float scale, int flags, int batch,
                 hipStream_t st) {
  dim3 grd((N + 63) / 64, (M + 63) / 64, batch);
  gemm_kernel<<<grd, 256, 0, st>>>(A, lda, sA, B, ldb, sB, bias, sBias,
                                   Cf, Cb, ldc, sC, M, N, K, scale, flags);
}

static void pack(const float* s, unsigned short* d, int n, hipStream_t st) {
  pack_bf16_kernel<<<(n + 255) / 256, 256, 0, st>>>(s, d, n);
}

extern "C" void kernel_launch(void* const* d_in, const int* in_sizes, int n_in,
                              void* d_out, int out_size, void* d_ws, size_t ws_size,
                              hipStream_t stream) {
  const float* tgt  = (const float*)d_in[0];
  const float* mem  = (const float*)d_in[1];
  const float* qcf  = (const float*)d_in[2];
  const float* kcf  = (const float*)d_in[3];
  const float* pos  = (const float*)d_in[4];
  const float* qse  = (const float*)d_in[5];
  const float* qc_w = (const float*)d_in[6];  const float* qc_b = (const float*)d_in[7];
  const float* kc_w = (const float*)d_in[8];  const float* kc_b = (const float*)d_in[9];
  const float* vp_w = (const float*)d_in[10]; const float* vp_b = (const float*)d_in[11];
  const float* kp_w = (const float*)d_in[12]; const float* kp_b = (const float*)d_in[13];
  const float* qs_w = (const float*)d_in[14]; const float* qs_b = (const float*)d_in[15];
  const float* aq_w = (const float*)d_in[16]; const float* aq_b = (const float*)d_in[17];
  const float* ak_w = (const float*)d_in[18]; const float* ak_b = (const float*)d_in[19];
  const float* av_w = (const float*)d_in[20]; const float* av_b = (const float*)d_in[21];
  const float* ao_w = (const float*)d_in[22]; const float* ao_b = (const float*)d_in[23];
  const float* f1_w = (const float*)d_in[24]; const float* f1_b = (const float*)d_in[25];
  const float* f2_w = (const float*)d_in[26]; const float* f2_b = (const float*)d_in[27];
  const float* relq = (const float*)d_in[28];
  const float* relk = (const float*)d_in[29];
  const float* relv = (const float*)d_in[30];
  const float* n2g  = (const float*)d_in[31]; const float* n2b = (const float*)d_in[32];
  const float* n3g  = (const float*)d_in[33]; const float* n3b = (const float*)d_in[34];
  float* outp = (float*)d_out;

  char* base = (char*)d_ws;
  size_t off = 0;
  auto A16 = [&](size_t e) { off = (off + 255) & ~(size_t)255; void* p = base + off; off += e * 2; return (unsigned short*)p; };
  auto A32 = [&](size_t e) { off = (off + 255) & ~(size_t)255; void* p = base + off; off += e * 4; return (float*)p; };

  unsigned short* tgt_bf = A16((size_t)MQ * DM);
  unsigned short* mem_bf = A16((size_t)MK * DM);
  unsigned short* pos_bf = A16((size_t)MK * DM);
  unsigned short* qse_bf = A16((size_t)MQ * DM);
  unsigned short* w_qc = A16(DM * DM);   unsigned short* w_kc = A16(DM * DM);
  unsigned short* w_vp = A16(DM * DM);   unsigned short* w_kp = A16(DM * DM);
  unsigned short* w_qs = A16(DM * DM);
  unsigned short* w_aq = A16(DM2 * DM2); unsigned short* w_ak = A16(DM2 * DM2);
  unsigned short* w_av = A16(DM * DM);   unsigned short* w_ao = A16(DM * DM);
  unsigned short* w_f1 = A16(DFFN * DM); unsigned short* w_f2 = A16(DM * DFFN);
  unsigned short* rt_q = A16(NHEAD * TBL * HDV);
  unsigned short* rt_k = A16(NHEAD * TBL * HDV);
  unsigned short* qbuf = A16((size_t)MQ * DM2);
  unsigned short* kbuf = A16((size_t)MK * DM2);
  unsigned short* vbuf = A16((size_t)MK * DM);
  unsigned short* qh_b = A16((size_t)MQ * DM2);
  unsigned short* kh_b = A16((size_t)MK * DM2);
  unsigned short* vh_b = A16((size_t)MK * DM);
  float*          sq_f = A32((size_t)MQ * NHEAD * TBL);
  float*          sk_f = A32((size_t)MK * NHEAD * TBL);
  unsigned short* ao_bf = A16((size_t)MQ * DM);
  float*          tg2_f = A32((size_t)MQ * DM);
  float*          x_f   = A32((size_t)MQ * DM);
  unsigned short* x_bf  = A16((size_t)MQ * DM);
  unsigned short* h_bf  = A16((size_t)MQ * DFFN);
  float*          ff_f  = A32((size_t)MQ * DM);

  // 1. convert to bf16
  pack(tgt, tgt_bf, MQ * DM, stream);
  pack(mem, mem_bf, MK * DM, stream);
  pack(pos, pos_bf, MK * DM, stream);
  pack(qse, qse_bf, MQ * DM, stream);
  pack(qc_w, w_qc, DM * DM, stream);   pack(kc_w, w_kc, DM * DM, stream);
  pack(vp_w, w_vp, DM * DM, stream);   pack(kp_w, w_kp, DM * DM, stream);
  pack(qs_w, w_qs, DM * DM, stream);
  pack(aq_w, w_aq, DM2 * DM2, stream); pack(ak_w, w_ak, DM2 * DM2, stream);
  pack(av_w, w_av, DM * DM, stream);   pack(ao_w, w_ao, DM * DM, stream);
  pack(f1_w, w_f1, DFFN * DM, stream); pack(f2_w, w_f2, DM * DFFN, stream);
  int nrel = NHEAD * TBL * HDV;
  rel_transpose_kernel<<<(nrel + 255) / 256, 256, 0, stream>>>(relq, rt_q);
  rel_transpose_kernel<<<(nrel + 255) / 256, 256, 0, stream>>>(relk, rt_k);

  // 2. projections, concat fused via column remap (flags 4 = lo, 8 = hi)
  gemm(tgt_bf, DM, 0, w_qc, DM, 0, qc_b, 0, nullptr, qbuf, DM2, 0, MQ, DM, DM, 1.f, 4, 1, stream);
  gemm(qse_bf, DM, 0, w_qs, DM, 0, qs_b, 0, nullptr, qbuf, DM2, 0, MQ, DM, DM, 1.f, 8, 1, stream);
  gemm(mem_bf, DM, 0, w_kc, DM, 0, kc_b, 0, nullptr, kbuf, DM2, 0, MK, DM, DM, 1.f, 4, 1, stream);
  gemm(pos_bf, DM, 0, w_kp, DM, 0, kp_b, 0, nullptr, kbuf, DM2, 0, MK, DM, DM, 1.f, 8, 1, stream);
  gemm(mem_bf, DM, 0, w_vp, DM, 0, vp_b, 0, nullptr, vbuf, DM, 0, MK, DM, DM, 1.f, 0, 1, stream);

  // 3. attention in-projections (qh scaled by HD_QK^-0.5 = 0.125)
  gemm(qbuf, DM2, 0, w_aq, DM2, 0, aq_b, 0, nullptr, qh_b, DM2, 0, MQ, DM2, DM2, 0.125f, 0, 1, stream);
  gemm(kbuf, DM2, 0, w_ak, DM2, 0, ak_b, 0, nullptr, kh_b, DM2, 0, MK, DM2, DM2, 1.f, 0, 1, stream);
  gemm(vbuf, DM, 0, w_av, DM, 0, av_b, 0, nullptr, vh_b, DM, 0, MK, DM, DM, 1.f, 0, 1, stream);

  // 4. RPE table projections, batched over heads (content halves, K=32)
  gemm(qh_b, DM2, HDQK, rt_q, HDV, (long long)TBL * HDV, nullptr, 0,
       sq_f, nullptr, NHEAD * TBL, TBL, MQ, TBL, HDV, 1.f, 0, NHEAD, stream);
  gemm(kh_b, DM2, HDQK, rt_k, HDV, (long long)TBL * HDV, nullptr, 0,
       sk_f, nullptr, NHEAD * TBL, TBL, MK, TBL, HDV, 1.f, 0, NHEAD, stream);

  // 5. fused flash attention + RPE + rel-value
  attn_kernel<<<dim3(NQ / 64, BSZ * NHEAD), 128, 0, stream>>>(
      qh_b, kh_b, vh_b, sq_f, sk_f, qcf, kcf, relv, ao_bf);

  // 6. out-proj, residual + LN, FFN, residual + LN
  gemm(ao_bf, DM, 0, w_ao, DM, 0, ao_b, 0, tg2_f, nullptr, DM, 0, MQ, DM, DM, 1.f, 0, 1, stream);
  add_ln_kernel<<<MQ / 8, 256, 0, stream>>>(tgt, tg2_f, n2g, n2b, x_f, x_bf, MQ);
  gemm(x_bf, DM, 0, w_f1, DM, 0, f1_b, 0, nullptr, h_bf, DFFN, 0, MQ, DFFN, DM, 1.f, 1, 1, stream);
  gemm(h_bf, DFFN, 0, w_f2, DFFN, 0, f2_b, 0, ff_f, nullptr, DM, 0, MQ, DM, DFFN, 1.f, 0, 1, stream);
  add_ln_kernel<<<MQ / 8, 256, 0, stream>>>(x_f, ff_f, n3g, n3b, outp, nullptr, MQ);
}
